// DyConvSelfAtt_8907762172170
// MI455X (gfx1250) — matrix operand order repaired and checked by reading, not compiled, not run on hardware
//
#include <hip/hip_runtime.h>
#include <hip/hip_bf16.h>

// ---------------------------------------------------------------------------
// DyConvSelfAtt for MI455X (gfx1250, wave32, WMMA).
//
// Pipeline (all f32 in/out, bf16 WMMA internally with f32 accumulation):
//   K_im2col : x -> bf16 im2col [200704 x 160]   (K=3*7*7=147 padded to 160)
//   K_prep   : bf16 weight repacks (fragment-friendly), zero padded-q buffer
//   K_qv     : WMMA implicit GEMM -> q (bf16, zero-padded 118x118) and v (f32)
//   K_attn   : FUSED kmap-conv + a-reduction. Per (b,i): for each 16-position
//              spatial tile and each qc: 5x v_wmma (K=160) produce the kmap
//              tile (16 pos x 16 v) -> +bk -> bf16 -> LDS -> re-fragment as
//              A2[v][qc,pos], gather shifted q as B2[qc,pos][16 shifts],
//              one v_wmma accumulates a[16v x 16s] across ALL spatial tiles.
//              kmap (308 MB if materialized) never touches HBM.
//   K_reduceA: deterministic sum of per-chunk partials of a.
//   K_out    : tiny 4x4 dynamic conv v (*) a -> o [16,3,58,58] (82 MFLOP).
//
// Roofline: 25 GFLOP total; fusion removes ~620 MB of kmap HBM traffic, the
// 64 MB im2col lives in the 192 MB L2 across K_qv/K_attn.
// Workspace: ~91 MB.
// ---------------------------------------------------------------------------

typedef __bf16 bf16;
typedef bf16  v16bf __attribute__((ext_vector_type(16)));
typedef bf16  v8bf  __attribute__((ext_vector_type(8)));
typedef float v8f   __attribute__((ext_vector_type(8)));

#define NB   16          // batch
#define IC   3           // input channels
#define QC   8           // q channels
#define VC   16          // v channels
#define HW   224
#define OHW  112         // conv output 112x112
#define NPOS 12544       // 112*112
#define KR   147         // 3*7*7
#define KP   160         // padded K (5 steps of 32)
#define QPW  118         // padded q width (pad 3 each side)
#define QPN  13924       // 118*118
#define NCHUNK 56        // spatial chunks per (b,i) in K_attn
#define TPC    14        // tiles per chunk (56*14 = 784 = 12544/16)

// ---- workspace layout (bytes) ----
#define OFF_IM2COL 0ull                                   // 200704*160*2 = 64,225,280
#define OFF_QPAD   64225280ull                            // 384*13924*2  = 10,693,632
#define OFF_VMAP   74918912ull                            // 256*12544*4  = 12,845,056
#define OFF_A      87763968ull                            // 48*256*4     = 49,152
#define OFF_APART  87813120ull                            // 56*48*256*4  = 2,752,512
#define OFF_WQVB   90565632ull                            // 40*160*2     = 12,800
#define OFF_WKB    90578432ull                            // 3*8*16*160*2 = 122,880

// ===========================================================================
// K_im2col: x [16,3,224,224] f32 -> bf16 im2col rows [m=200704][k=160]
// ===========================================================================
__global__ void K_im2col(const float* __restrict__ x, bf16* __restrict__ col) {
  int idx = blockIdx.x * 256 + threadIdx.x;
  if (idx >= NB * NPOS * KP) return;
  int k = idx % KP;
  int m = idx / KP;
  float val = 0.f;
  if (k < KR) {
    int cin = k / 49, r = (k % 49) / 7, s = k % 7;
    int b = m / NPOS, rem = m % NPOS;
    int oy = rem / OHW, ox = rem % OHW;
    int iy = 2 * oy - 3 + r, ix = 2 * ox - 3 + s;
    if (iy >= 0 && iy < HW && ix >= 0 && ix < HW)
      val = x[(((size_t)(b * IC + cin)) * HW + iy) * HW + ix];
  }
  col[idx] = (bf16)val;
}

// ===========================================================================
// K_prep: repack weights to bf16 fragment-friendly layouts; zero q-pad buffer
//   wqvb[ch][k=160]  ch<24: Wq flat, 24..39: Wv           (B-fragment rows)
//   wkb [i][qc][v][k=160]                                  (B-fragment rows)
// ===========================================================================
__global__ void K_prep(const float* __restrict__ Wq, const float* __restrict__ Wv,
                       const float* __restrict__ Wk,
                       bf16* __restrict__ wqvb, bf16* __restrict__ wkb,
                       bf16* __restrict__ qpad) {
  int idx = blockIdx.x * 256 + threadIdx.x;
  const int N1 = 40 * KP;                 // 6400
  const int N2 = N1 + IC * QC * VC * KP;  // +61440
  const int N3 = N2 + IC * QC * NB * QPN; // + q_pad elems (384*13924)
  if (idx < N1) {
    int ch = idx / KP, k = idx % KP;
    float v = 0.f;
    if (k < KR) v = (ch < 24) ? Wq[ch * KR + k] : Wv[(ch - 24) * KR + k];
    wqvb[idx] = (bf16)v;
  } else if (idx < N2) {
    int t = idx - N1;
    int k = t % KP;
    int r = t / KP;
    int v = r % 16; r /= 16;
    int qc = r % 8;
    int i  = r / 8;
    float val = 0.f;
    if (k < KR) val = Wk[(((size_t)((i * VC + v) * QC + qc)) * KR) + k];
    wkb[t] = (bf16)val;
  } else if (idx < N3) {
    qpad[idx - N2] = (bf16)0.f;
  }
}

// ===========================================================================
// K_qv: WMMA implicit GEMM. M = b*112*112 (tiles of 16, row-aligned since
// 112%16==0), N = 40 channels (24 q + 16 v), K = 160. One wave per 16x16 tile.
// Epilogue: +bias; q -> bf16 zero-padded [b,i,qc,118,118]; v -> f32.
// ===========================================================================
__global__ void __launch_bounds__(32)
K_qv(const bf16* __restrict__ col, const bf16* __restrict__ wqvb,
     const float* __restrict__ bq, const float* __restrict__ bv,
     bf16* __restrict__ qpad, float* __restrict__ vmap) {
  int lane = threadIdx.x;
  int hi = lane >> 4;        // 0/1: which K-half of the wave
  int lo = lane & 15;
  int mb = blockIdx.x * 16;  // 16 consecutive spatial positions, same row
  int ntile = blockIdx.y;    // 0..2
  int ch = ntile * 16 + lo;

  // A fragment rows: M = lo (im2col row mb+lo). lane<16 holds
  // K = e (0..15); lanes>=16 add +16  -> two contiguous 8-elem runs.
  const bf16* arow = col + (size_t)(mb + lo) * KP + (hi << 4);
  // B fragment: N = lo, K = e + 16*hi (linear within half) -> one 32B load.
  const bf16* brow = wqvb + ch * KP + (hi << 4);

  v8f c = {};
#pragma unroll
  for (int ks = 0; ks < 5; ++ks) {
    v8bf alo = *(const v8bf*)(arow + ks * 32);
    v8bf ahi = *(const v8bf*)(arow + ks * 32 + 8);
    v16bf af;
#pragma unroll
    for (int e = 0; e < 8; ++e) { af[e] = alo[e]; af[e + 8] = ahi[e]; }
    v16bf bfr = *(const v16bf*)(brow + ks * 32);
    c = __builtin_amdgcn_wmma_f32_16x16x32_bf16(false, af, false, bfr,
                                                (short)0, c, false, false);
  }

  // D layout: M = r + 8*hi (spatial), N = lo (channel)
  int b = mb / NPOS;
  int rem = mb % NPOS;
  int oy = rem / OHW, ox0 = rem % OHW;
  if (ch < 24) {
    float bias = bq[ch];
    int i_ = ch >> 3, qc = ch & 7;
    bf16* qrow = qpad + ((size_t)((b * IC + i_) * QC + qc)) * QPN
               + (oy + 3) * QPW + (ox0 + 3);
#pragma unroll
    for (int r = 0; r < 8; ++r) qrow[r + hi * 8] = (bf16)(c[r] + bias);
  } else if (ch < 40) {
    int vch = ch - 24;
    float bias = bv[vch];
    float* vrow = vmap + ((size_t)(b * VC + vch)) * NPOS + oy * OHW + ox0;
#pragma unroll
    for (int r = 0; r < 8; ++r) vrow[r + hi * 8] = c[r] + bias;
  }
}

// ===========================================================================
// K_attn: fused kmap-conv + a-reduction. grid = (56 chunks, 48 (b,i)), 1 wave.
// a[v,s] = sum_{qc,j} (kmap_conv[v,qc,j]+bk) * q[qc, j+s], 16 shifts s.
// ===========================================================================
__global__ void __launch_bounds__(32)
K_attn(const bf16* __restrict__ col, const bf16* __restrict__ wkb,
       const float* __restrict__ bk, const bf16* __restrict__ qpad,
       float* __restrict__ apart) {
  __shared__ __align__(16) bf16 kls[16][32];  // [v][K = sub*16 + pos]

  int lane = threadIdx.x;
  int hi = lane >> 4;
  int lo = lane & 15;
  int bi = blockIdx.y;              // b*3+i
  int b = bi / IC, i = bi % IC;
  int chunk = blockIdx.x;

  const bf16* qbase = qpad + (size_t)bi * QC * QPN;
  // B2 column = shift index s: (sy,sx) = (2*(s/4)-3, 2*(s%4)-3)
  int sy = 2 * (lo >> 2) - 3, sx = 2 * (lo & 3) - 3;

  v8f acc = {};  // a accumulator: 16 v x 16 s, lives across the whole chunk

  for (int tt = 0; tt < TPC; ++tt) {
    int t = chunk * TPC + tt;
    int oy = t / 7, ox0 = (t % 7) * 16;  // 7 tiles per row, 112 rows
    int rowm = b * NPOS + oy * OHW + ox0 + lo;

    // Cache the 5 im2col A-fragments of this tile (shared by all 8 qc GEMMs)
    const bf16* arow = col + (size_t)rowm * KP + (hi << 4);
    v16bf A1[5];
#pragma unroll
    for (int ks = 0; ks < 5; ++ks) {
      v8bf alo = *(const v8bf*)(arow + ks * 32);
      v8bf ahi = *(const v8bf*)(arow + ks * 32 + 8);
#pragma unroll
      for (int e = 0; e < 8; ++e) { A1[ks][e] = alo[e]; A1[ks][e + 8] = ahi[e]; }
    }
    if (tt + 1 < TPC)  // stream next tile's im2col rows (global_prefetch_b8)
      __builtin_prefetch(col + (size_t)(rowm + 16) * KP, 0, 1);

#pragma unroll
    for (int qp = 0; qp < 4; ++qp) {       // qc pairs -> one a-GEMM each
#pragma unroll
      for (int sub = 0; sub < 2; ++sub) {
        int qc = qp * 2 + sub;
        // kmap implicit GEMM: D[pos][v], A=im2col, B=Wk rows (N = v = lo)
        const bf16* brow = wkb + (size_t)(((i * QC + qc) * 16) + lo) * KP
                         + (hi << 4);
        v8f c1 = {};
#pragma unroll
        for (int ks = 0; ks < 5; ++ks) {
          v16bf bfr = *(const v16bf*)(brow + ks * 32);
          c1 = __builtin_amdgcn_wmma_f32_16x16x32_bf16(false, A1[ks], false, bfr,
                                                       (short)0, c1, false, false);
        }
        // D: M = pos = r + 8*hi, N = v = lo. Stage (+bias) to LDS as bf16.
        float bkv = bk[(i * VC + lo) * QC + qc];
#pragma unroll
        for (int r = 0; r < 8; ++r)
          kls[lo][sub * 16 + r + hi * 8] = (bf16)(c1[r] + bkv);
      }
      __syncthreads();

      // A2: rows = v (lo), K = sub*16+pos; two contiguous 8-elem LDS runs
      v8bf l0 = *(const v8bf*)&kls[lo][hi << 3];
      v8bf l1 = *(const v8bf*)&kls[lo][(hi << 3) + 16];
      v16bf a2;
#pragma unroll
      for (int e = 0; e < 8; ++e) { a2[e] = l0[e]; a2[e + 8] = l1[e]; }

      // B2: [K][s] = q[qc(K), (oy,ox0+pos(K)) + (sy,sx)] from padded q
      v16bf b2;
      int rowbase = (oy + 3 + sy) * QPW + (ox0 + 3 + sx);
#pragma unroll
      for (int e = 0; e < 16; ++e) {
        int K = ((e & 8) ? 16 : 0) + (hi << 3) + (e & 7);
        int qc2 = qp * 2 + (K >> 4);
        int pos = K & 15;
        b2[e] = qbase[(size_t)qc2 * QPN + rowbase + pos];
      }
      acc = __builtin_amdgcn_wmma_f32_16x16x32_bf16(false, a2, false, b2,
                                                    (short)0, acc, false, false);
      __syncthreads();  // LDS reused by next qc pair
    }
  }

  // Commit per-chunk partial (deterministic; reduced by K_reduceA).
  // D: M = v = r + 8*hi, N = s = lo.
  float* pp = apart + (size_t)(bi * NCHUNK + chunk) * 256;
#pragma unroll
  for (int r = 0; r < 8; ++r) pp[(r + hi * 8) * 16 + lo] = acc[r];
}

// ===========================================================================
// K_reduceA: a[bi][v][s] = sum over 56 chunk partials (deterministic order)
// ===========================================================================
__global__ void K_reduceA(const float* __restrict__ apart, float* __restrict__ a) {
  int idx = blockIdx.x * 256 + threadIdx.x;
  if (idx >= 48 * 256) return;
  int bi = idx >> 8, vs = idx & 255;
  float s = 0.f;
  for (int c = 0; c < NCHUNK; ++c) s += apart[(size_t)(bi * NCHUNK + c) * 256 + vs];
  a[idx] = s;
}

// ===========================================================================
// K_out: o[b,i,oy,ox] = sum_{v,kh,kw} v[b,v,2oy-3+kh,2ox-3+kw]*a[b,i,v,kh,kw]
// 82 MFLOP -> plain f32 (0.3% of total work).
// ===========================================================================
__global__ void K_out(const float* __restrict__ vmap, const float* __restrict__ a,
                      float* __restrict__ out) {
  int idx = blockIdx.x * 256 + threadIdx.x;
  if (idx >= NB * IC * 58 * 58) return;
  int ox = idx % 58; int t = idx / 58;
  int oy = t % 58;   t /= 58;
  int i = t % IC;    int b = t / IC;
  const float* ab = a + (size_t)(b * IC + i) * 256;
  const float* vb = vmap + (size_t)b * VC * NPOS;
  float acc = 0.f;
  for (int v = 0; v < VC; ++v) {
    const float* vv = vb + v * NPOS;
    const float* av = ab + v * 16;
#pragma unroll
    for (int kh = 0; kh < 4; ++kh) {
      int iy = 2 * oy - 3 + kh;
      if (iy < 0 || iy >= OHW) continue;
#pragma unroll
      for (int kw = 0; kw < 4; ++kw) {
        int ix = 2 * ox - 3 + kw;
        if (ix < 0 || ix >= OHW) continue;
        acc += vv[iy * OHW + ix] * av[kh * 4 + kw];
      }
    }
  }
  out[idx] = acc;
}

// ===========================================================================
extern "C" void kernel_launch(void* const* d_in, const int* in_sizes, int n_in,
                              void* d_out, int out_size, void* d_ws, size_t ws_size,
                              hipStream_t stream) {
  const float* x  = (const float*)d_in[0];
  const float* Wq = (const float*)d_in[1];
  const float* bq = (const float*)d_in[2];
  const float* Wk = (const float*)d_in[3];
  const float* bk = (const float*)d_in[4];
  const float* Wv = (const float*)d_in[5];
  const float* bv = (const float*)d_in[6];
  float* out = (float*)d_out;

  char* ws = (char*)d_ws;
  bf16*  im2col = (bf16*)(ws + OFF_IM2COL);
  bf16*  qpad   = (bf16*)(ws + OFF_QPAD);
  float* vmap   = (float*)(ws + OFF_VMAP);
  float* a      = (float*)(ws + OFF_A);
  float* apart  = (float*)(ws + OFF_APART);
  bf16*  wqvb   = (bf16*)(ws + OFF_WQVB);
  bf16*  wkb    = (bf16*)(ws + OFF_WKB);

  {
    int n = NB * NPOS * KP;
    K_im2col<<<(n + 255) / 256, 256, 0, stream>>>(x, im2col);
  }
  {
    int n = 40 * KP + IC * QC * VC * KP + IC * QC * NB * QPN;
    K_prep<<<(n + 255) / 256, 256, 0, stream>>>(Wq, Wv, Wk, wqvb, wkb, qpad);
  }
  K_qv<<<dim3(NPOS, 3), 32, 0, stream>>>(im2col, wqvb, bq, bv, qpad, vmap);
  K_attn<<<dim3(NCHUNK, NB * IC), 32, 0, stream>>>(im2col, wkb, bk, qpad, apart);
  K_reduceA<<<48, 256, 0, stream>>>(apart, a);
  {
    int n = NB * IC * 58 * 58;
    K_out<<<(n + 255) / 256, 256, 0, stream>>>(vmap, a, out);
  }
}
